// DAttentionBaseline_75376676044867
// MI455X (gfx1250) — matrix operand order repaired and checked by reading, not compiled, not run on hardware
//
#include <hip/hip_runtime.h>

typedef __attribute__((ext_vector_type(16))) _Float16 v16h;
typedef __attribute__((ext_vector_type(8)))  _Float16 v8h;
typedef __attribute__((ext_vector_type(8)))  float    v8f;

constexpr int   kB = 4, kC = 512, kH = 32, kW = 32, kHW = 1024;
constexpr int   kHEADS = 8, kHC = 64, kG = 4, kGC = 128, kGH = 2;
constexpr int   kR = 63;                 // rpe table spatial = 2*H-1
constexpr float kSCALE = 0.125f;         // 64^-0.5
constexpr float kEPS = 1e-5f;
constexpr int   kNELEM = kB * kC * kHW;  // 2,097,152

// ---------------------------------------------------------------------------
// WMMA fragment helpers (CDNA5 wave32, 16x16x32 f16 -> f32), K-contiguous
// sources so each lane's halves are vector loads.
// A 16x32 (MxK): lane: m = lane&15, half = lane>>4
//   halves 0..7  = A[m][half*8 + 0..7]        (16B load)
//   halves 8..15 = A[m][16 + half*8 + 0..7]   (16B load)
// B 32x16 (KxN): lane: n = lane&15, half = lane>>4
//   halves 0..7  = B[half*8 + 0..7][n], halves 8..15 = B[16 + half*8 + 0..7][n]
// C/D 16x16 f32: vgpr r, lane: D[m = r + 8*(lane>>4)][n = lane&15]
// ---------------------------------------------------------------------------
__device__ inline v16h load_frag_a_kc(const _Float16* __restrict__ src, int ldm) {
  const int lane = threadIdx.x & 31;
  const int m = lane & 15, half = lane >> 4;
  const _Float16* p = src + (size_t)m * ldm + half * 8;
  v8h lo = *(const v8h*)p;
  v8h hi = *(const v8h*)(p + 16);
  return __builtin_shufflevector(lo, hi, 0, 1, 2, 3, 4, 5, 6, 7, 8, 9, 10, 11, 12,
                                 13, 14, 15);
}

__device__ inline v16h load_frag_b_kc(const _Float16* __restrict__ src, int ldn) {
  const int lane = threadIdx.x & 31;
  const int nn = lane & 15, half = lane >> 4;
  const _Float16* p = src + (size_t)nn * ldn + half * 8;
  v8h lo = *(const v8h*)p;
  v8h hi = *(const v8h*)(p + 16);
  return __builtin_shufflevector(lo, hi, 0, 1, 2, 3, 4, 5, 6, 7, 8, 9, 10, 11, 12,
                                 13, 14, 15);
}

__device__ inline v8f wmma_f16(v16h a, v16h b, v8f c) {
  return __builtin_amdgcn_wmma_f32_16x16x32_f16(false, a, false, b, (short)0, c,
                                                false, false);
}

__device__ inline int iclamp(int v, int lo, int hi) {
  return v < lo ? lo : (v > hi ? hi : v);
}

// ---------------------------------------------------------------------------
// fp32 -> fp16 conversion (weights, row-major kept)
// ---------------------------------------------------------------------------
__global__ void cvt_f16_kernel(const float* __restrict__ s, _Float16* __restrict__ d,
                               int n) {
  int i = blockIdx.x * blockDim.x + threadIdx.x;
  if (i < n) d[i] = (_Float16)s[i];
}

// x [B,C,HW] f32 -> xT [B,HW,C] f16 (output-indexed, coalesced writes)
__global__ void cvt_xT_kernel(const float* __restrict__ s, _Float16* __restrict__ d) {
  int i = blockIdx.x * blockDim.x + threadIdx.x;
  const int c = i & (kC - 1);
  const int rest = i >> 9;   // /kC
  const int p = rest & (kHW - 1);
  const int b = rest >> 10;  // /kHW
  d[i] = (_Float16)s[((size_t)b * kC + c) * kHW + p];
}

// ---------------------------------------------------------------------------
// WMMA GEMM:  acc[m][n] = sum_k Wm[m][k] * InT[z][n][k] + bias[m]
// block = 128 (4 waves); each wave: 16(M) x 64(N) via 4 accumulators.
// MODE 0: outF [M,N] f32 + outH [N,M] f16 (q)
// MODE 1: outH [N,M] f16                  (k)
// MODE 2: outH [M,N] f16                  (v)
// MODE 3: outF [M,N] f32                  (y)
// ---------------------------------------------------------------------------
template <int MODE>
__global__ __launch_bounds__(128, 1) void gemm_wmma_kernel(
    const _Float16* __restrict__ Wm, const _Float16* __restrict__ InT,
    const float* __restrict__ bias, float* __restrict__ outF,
    _Float16* __restrict__ outH, int M, int N, int K) {
  const int lane = threadIdx.x & 31;
  const int wid  = threadIdx.x >> 5;
  const int m0 = blockIdx.y * 16;
  const int nB = (blockIdx.x * (blockDim.x >> 5) + wid) * 64;
  const size_t iofs = (size_t)blockIdx.z * N * K;
  const size_t oofs = (size_t)blockIdx.z * M * N;

  const v8f vz = {0.f, 0.f, 0.f, 0.f, 0.f, 0.f, 0.f, 0.f};
  v8f acc[4] = {vz, vz, vz, vz};
  for (int k0 = 0; k0 < K; k0 += 32) {
    const v16h a = load_frag_a_kc(Wm + (size_t)m0 * K + k0, K);
#pragma unroll
    for (int t = 0; t < 4; ++t) {
      const v16h b = load_frag_b_kc(InT + iofs + (size_t)(nB + t * 16) * K + k0, K);
      acc[t] = wmma_f16(a, b, acc[t]);
    }
  }
  const int col = lane & 15, halfw = lane >> 4;
#pragma unroll
  for (int t = 0; t < 4; ++t) {
#pragma unroll
    for (int r = 0; r < 8; ++r) {
      const int m = m0 + r + 8 * halfw;
      const int n = nB + t * 16 + col;
      const float v = acc[t][r] + bias[m];
      if constexpr (MODE == 0) {
        outF[oofs + (size_t)m * N + n] = v;
        outH[oofs + (size_t)n * M + m] = (_Float16)v;
      } else if constexpr (MODE == 1) {
        outH[oofs + (size_t)n * M + m] = (_Float16)v;
      } else if constexpr (MODE == 2) {
        outH[oofs + (size_t)m * N + n] = (_Float16)v;
      } else {
        outF[oofs + (size_t)m * N + n] = v;
      }
    }
  }
}

// ---------------------------------------------------------------------------
// Offset network: depthwise 3x3 -> channel LayerNorm -> GELU -> 1x1 conv to 2
// -> add reference points -> clip -> pos.  One block per (bg, pixel).
// ---------------------------------------------------------------------------
__global__ __launch_bounds__(kGC) void offset_net_kernel(
    const float* __restrict__ q32, const float* __restrict__ dw_w,
    const float* __restrict__ dw_b, const float* __restrict__ ln_w,
    const float* __restrict__ ln_b, const float* __restrict__ pw_w,
    float* __restrict__ pos, float* __restrict__ pos_out,
    float* __restrict__ ref_out) {
  const int p  = blockIdx.x;   // 0..1023
  const int bg = blockIdx.y;   // 0..15
  const int c  = threadIdx.x;  // 0..127
  const int b = bg / kG, g = bg % kG;
  const int y = p / kW, x = p % kW;

  const float* qc = q32 + ((size_t)b * kC + g * kGC + c) * kHW;
  float o = dw_b[c];
#pragma unroll
  for (int ky = 0; ky < 3; ++ky) {
    const int yy = y + ky - 1;
    if (yy < 0 || yy >= kH) continue;
#pragma unroll
    for (int kx = 0; kx < 3; ++kx) {
      const int xx = x + kx - 1;
      if (xx < 0 || xx >= kW) continue;
      o += qc[yy * kW + xx] * dw_w[c * 9 + ky * 3 + kx];
    }
  }

  __shared__ float buf[kGC];
  buf[c] = o;
  __syncthreads();
  for (int s = kGC / 2; s > 0; s >>= 1) {
    if (c < s) buf[c] += buf[c + s];
    __syncthreads();
  }
  const float mu = buf[0] * (1.0f / kGC);
  __syncthreads();
  const float d0 = o - mu;
  buf[c] = d0 * d0;
  __syncthreads();
  for (int s = kGC / 2; s > 0; s >>= 1) {
    if (c < s) buf[c] += buf[c + s];
    __syncthreads();
  }
  const float var = buf[0] * (1.0f / kGC);
  __syncthreads();

  o = d0 * rsqrtf(var + kEPS) * ln_w[c] + ln_b[c];
  o = 0.5f * o * (1.0f + erff(o * 0.70710678118654752f));  // exact GELU

  buf[c] = o * pw_w[c];
  __syncthreads();
  for (int s = kGC / 2; s > 0; s >>= 1) {
    if (c < s) buf[c] += buf[c + s];
    __syncthreads();
  }
  const float off_y = buf[0];
  __syncthreads();
  buf[c] = o * pw_w[kGC + c];
  __syncthreads();
  for (int s = kGC / 2; s > 0; s >>= 1) {
    if (c < s) buf[c] += buf[c + s];
    __syncthreads();
  }
  const float off_x = buf[0];

  if (c == 0) {
    const float ry = ((y + 0.5f) / (kH - 1.0f)) * 2.0f - 1.0f;
    const float rx = ((x + 0.5f) / (kW - 1.0f)) * 2.0f - 1.0f;
    const float py = fminf(fmaxf(off_y + ry, -1.f), 1.f);
    const float px = fminf(fmaxf(off_x + rx, -1.f), 1.f);
    const size_t idx = ((size_t)bg * kHW + p) * 2;
    pos[idx] = py;      pos[idx + 1] = px;
    pos_out[idx] = py;  pos_out[idx + 1] = px;
    ref_out[idx] = ry;  ref_out[idx + 1] = rx;
  }
}

// ---------------------------------------------------------------------------
// Deformable bilinear sampling of x at pos -> xsT [B,HW,C] f16 (coalesced).
// ---------------------------------------------------------------------------
__global__ __launch_bounds__(kGC) void sample_kernel(const float* __restrict__ x,
                                                     const float* __restrict__ pos,
                                                     _Float16* __restrict__ xsT) {
  const int p = blockIdx.x, bg = blockIdx.y, c = threadIdx.x;
  const int b = bg / kG, g = bg % kG;
  const size_t pidx = ((size_t)bg * kHW + p) * 2;
  const float py = pos[pidx], px = pos[pidx + 1];
  const float gx = (px + 1.f) * 0.5f * (kW - 1);
  const float gy = (py + 1.f) * 0.5f * (kH - 1);
  const float fx = floorf(gx), fy = floorf(gy);
  const float wx = gx - fx, wy = gy - fy;
  const int x0 = iclamp((int)fx, 0, kW - 1);
  const int x1 = iclamp((int)fx + 1, 0, kW - 1);
  const int y0 = iclamp((int)fy, 0, kH - 1);
  const int y1 = iclamp((int)fy + 1, 0, kH - 1);
  const float* xc = x + ((size_t)b * kC + g * kGC + c) * kHW;
  const float v00 = xc[y0 * kW + x0], v01 = xc[y0 * kW + x1];
  const float v10 = xc[y1 * kW + x0], v11 = xc[y1 * kW + x1];
  const float v = v00 * (1.f - wx) * (1.f - wy) + v01 * wx * (1.f - wy) +
                  v10 * (1.f - wx) * wy + v11 * wx * wy;
  xsT[((size_t)b * kHW + p) * kC + g * kGC + c] = (_Float16)v;
}

// ---------------------------------------------------------------------------
// Bilinear rpe lookup from LDS.
// ---------------------------------------------------------------------------
__device__ inline float bilin_rpe(const float* __restrict__ rpe_s, float qy, float qx,
                                  float py, float px) {
  const float dy = (qy - py) * 0.5f, dx = (qx - px) * 0.5f;
  const float gx = (dx + 1.f) * 0.5f * (kR - 1);
  const float gy = (dy + 1.f) * 0.5f * (kR - 1);
  const float fx = floorf(gx), fy = floorf(gy);
  const float wx = gx - fx, wy = gy - fy;
  const int x0 = iclamp((int)fx, 0, kR - 1);
  const int x1 = iclamp((int)fx + 1, 0, kR - 1);
  const int y0 = iclamp((int)fy, 0, kR - 1);
  const int y1 = iclamp((int)fy + 1, 0, kR - 1);
  const float v00 = rpe_s[y0 * kR + x0], v01 = rpe_s[y0 * kR + x1];
  const float v10 = rpe_s[y1 * kR + x0], v11 = rpe_s[y1 * kR + x1];
  return v00 * (1.f - wx) * (1.f - wy) + v01 * wx * (1.f - wy) +
         v10 * (1.f - wx) * wy + v11 * wx * wy;
}

// ---------------------------------------------------------------------------
// Fused flash attention with deformable rpe bias.
// grid = (HW/16, B*HEADS), block = 32 (one wave, 16 query rows).
// qT/kT [B,HW,C] (K-contiguous), v [B,C,HW].  Output oT [B,HW,C].
// __launch_bounds__(32,1): single wave per block, let the allocator keep
// q fragments + 4 accumulators + softmax state resident (no scratch spills).
// ---------------------------------------------------------------------------
__global__ __launch_bounds__(32, 1) void attn_kernel(
    const _Float16* __restrict__ qT, const _Float16* __restrict__ kT,
    const _Float16* __restrict__ vh, const float* __restrict__ pos,
    const float* __restrict__ rpe, _Float16* __restrict__ oT) {
  __shared__ float    rpe_s[kR * kR];
  __shared__ _Float16 p_s[16 * 32];

  const int lane = threadIdx.x & 31;
  const int bh = blockIdx.y;
  const int b = bh / kHEADS, h = bh % kHEADS;
  const int bg = b * kG + h / kGH;
  const int m0 = blockIdx.x * 16;
  const int col = lane & 15, halfw = lane >> 4;

  for (int i = threadIdx.x; i < kR * kR; i += 32) rpe_s[i] = rpe[h * kR * kR + i];
  __syncthreads();

  const _Float16* qp = qT + ((size_t)b * kHW + m0) * kC + h * kHC;
  const _Float16* kp = kT + (size_t)b * kHW * kC + h * kHC;
  const _Float16* vp = vh + ((size_t)b * kC + h * kHC) * kHW;

  // q A-fragments (K-contiguous, fixed across all chunks): K = 64 in two steps
  const v16h aq0 = load_frag_a_kc(qp, kC);
  const v16h aq1 = load_frag_a_kc(qp + 32, kC);

  const v8f vz = {0.f, 0.f, 0.f, 0.f, 0.f, 0.f, 0.f, 0.f};
  v8f acc[4] = {vz, vz, vz, vz};
  float Mrow[8], Lrow[8];
#pragma unroll
  for (int r = 0; r < 8; ++r) { Mrow[r] = -1e30f; Lrow[r] = 0.f; }

  for (int nc = 0; nc < kHW; nc += 32) {
    // ---- S tiles: two 16-wide subtiles, K = 64 ----
    v8f s0 = vz, s1 = vz;
    {
      v16h bk = load_frag_b_kc(kp + (size_t)nc * kC, kC);
      s0 = wmma_f16(aq0, bk, s0);
      bk = load_frag_b_kc(kp + (size_t)nc * kC + 32, kC);
      s0 = wmma_f16(aq1, bk, s0);
      bk = load_frag_b_kc(kp + (size_t)(nc + 16) * kC, kC);
      s1 = wmma_f16(aq0, bk, s1);
      bk = load_frag_b_kc(kp + (size_t)(nc + 16) * kC + 32, kC);
      s1 = wmma_f16(aq1, bk, s1);
    }
    // ---- scale + rpe bias (LDS bilinear); query coords recomputed inline ----
    const size_t p0i = ((size_t)bg * kHW + nc + col) * 2;
    const float py0 = pos[p0i], px0 = pos[p0i + 1];
    const float py1 = pos[p0i + 32], px1 = pos[p0i + 33];
#pragma unroll
    for (int r = 0; r < 8; ++r) {
      const int m = m0 + r + 8 * halfw;
      const float qy = (float)(m >> 5) * (2.0f / (kH - 1)) - 1.0f;
      const float qx = (float)(m & 31) * (2.0f / (kW - 1)) - 1.0f;
      s0[r] = s0[r] * kSCALE + bilin_rpe(rpe_s, qy, qx, py0, px0);
      s1[r] = s1[r] * kSCALE + bilin_rpe(rpe_s, qy, qx, py1, px1);
    }
    // ---- online softmax ----
#pragma unroll
    for (int r = 0; r < 8; ++r) {
      float mval = fmaxf(s0[r], s1[r]);
#pragma unroll
      for (int msk = 1; msk < 16; msk <<= 1)
        mval = fmaxf(mval, __shfl_xor(mval, msk, 16));
      const float newM = fmaxf(Mrow[r], mval);
      const float alpha = __expf(Mrow[r] - newM);
      const float e0 = __expf(s0[r] - newM), e1 = __expf(s1[r] - newM);
      float rs = e0 + e1;
#pragma unroll
      for (int msk = 1; msk < 16; msk <<= 1) rs += __shfl_xor(rs, msk, 16);
      Lrow[r] = Lrow[r] * alpha + rs;
      Mrow[r] = newM;
      acc[0][r] *= alpha; acc[1][r] *= alpha; acc[2][r] *= alpha; acc[3][r] *= alpha;
      const int m = r + 8 * halfw;
      p_s[m * 32 + col]      = (_Float16)e0;
      p_s[m * 32 + 16 + col] = (_Float16)e1;
    }
    __syncthreads();
    // ---- O += P (16x32) * v^T (32x64) ----
    const v16h a2 = load_frag_a_kc(p_s, 32);
#pragma unroll
    for (int t = 0; t < 4; ++t) {
      const v16h b2 = load_frag_b_kc(vp + (size_t)t * 16 * kHW + nc, kHW);
      acc[t] = wmma_f16(a2, b2, acc[t]);
    }
    __syncthreads();
  }

  // ---- epilogue: normalize, store oT [B,HW,C] f16 ----
#pragma unroll
  for (int t = 0; t < 4; ++t) {
#pragma unroll
    for (int r = 0; r < 8; ++r) {
      const float v = acc[t][r] / Lrow[r];
      oT[((size_t)b * kHW + m0 + r + 8 * halfw) * kC + h * kHC + t * 16 + col] =
          (_Float16)v;
    }
  }
}

// ---------------------------------------------------------------------------
extern "C" void kernel_launch(void* const* d_in, const int* in_sizes, int n_in,
                              void* d_out, int out_size, void* d_ws, size_t ws_size,
                              hipStream_t stream) {
  const float* x    = (const float*)d_in[0];
  const float* wq   = (const float*)d_in[1];
  const float* bq   = (const float*)d_in[2];
  const float* wk   = (const float*)d_in[3];
  const float* bk   = (const float*)d_in[4];
  const float* wv   = (const float*)d_in[5];
  const float* bv   = (const float*)d_in[6];
  const float* wo   = (const float*)d_in[7];
  const float* bo   = (const float*)d_in[8];
  const float* dw_w = (const float*)d_in[9];
  const float* dw_b = (const float*)d_in[10];
  const float* ln_w = (const float*)d_in[11];
  const float* ln_b = (const float*)d_in[12];
  const float* pw_w = (const float*)d_in[13];
  const float* rpe  = (const float*)d_in[14];

  char* wsb = (char*)d_ws;
  size_t off = 0;
  auto carve = [&](size_t bytes) {
    void* p = wsb + off;
    off += (bytes + 255) & ~(size_t)255;
    return p;
  };
  float*    q32 = (float*)carve((size_t)kNELEM * 4);     // q [B,C,HW] f32
  _Float16* qT  = (_Float16*)carve((size_t)kNELEM * 2);  // q^T [B,HW,C]
  _Float16* xT  = (_Float16*)carve((size_t)kNELEM * 2);  // x^T [B,HW,C]
  _Float16* wqh = (_Float16*)carve((size_t)kC * kC * 2);
  _Float16* wkh = (_Float16*)carve((size_t)kC * kC * 2);
  _Float16* wvh = (_Float16*)carve((size_t)kC * kC * 2);
  _Float16* woh = (_Float16*)carve((size_t)kC * kC * 2);
  _Float16* xsT = (_Float16*)carve((size_t)kNELEM * 2);  // xs^T [B,n,C]
  _Float16* kT  = (_Float16*)carve((size_t)kNELEM * 2);  // k^T [B,n,C]
  _Float16* vhd = (_Float16*)carve((size_t)kNELEM * 2);  // v [B,C,n]
  _Float16* oT  = (_Float16*)carve((size_t)kNELEM * 2);  // o^T [B,HW,C]
  float*    pos = (float*)carve((size_t)kB * kG * kHW * 2 * 4);

  float* y_out   = (float*)d_out;
  float* pos_out = y_out + kNELEM;
  float* ref_out = pos_out + (size_t)kB * kG * kHW * 2;

  // 1) conversions
  cvt_xT_kernel<<<kNELEM / 256, 256, 0, stream>>>(x, xT);
  const int wsz = kC * kC;
  cvt_f16_kernel<<<(wsz + 255) / 256, 256, 0, stream>>>(wq, wqh, wsz);
  cvt_f16_kernel<<<(wsz + 255) / 256, 256, 0, stream>>>(wk, wkh, wsz);
  cvt_f16_kernel<<<(wsz + 255) / 256, 256, 0, stream>>>(wv, wvh, wsz);
  cvt_f16_kernel<<<(wsz + 255) / 256, 256, 0, stream>>>(wo, woh, wsz);

  const dim3 gemm_grid(kHW / 256, kC / 16, kB);  // 4 waves x 64 N each

  // 2) q = wq * x + bq  (f32 [C,HW] + f16 [HW,C])
  gemm_wmma_kernel<0><<<gemm_grid, 128, 0, stream>>>(wqh, xT, bq, q32, qT, kC, kHW, kC);

  // 3) offset network -> pos (+ pos/ref output tails)
  offset_net_kernel<<<dim3(kHW, kB * kG), kGC, 0, stream>>>(q32, dw_w, dw_b, ln_w,
                                                            ln_b, pw_w, pos, pos_out,
                                                            ref_out);

  // 4) deformable sampling -> xs^T (f16)
  sample_kernel<<<dim3(kHW, kB * kG), kGC, 0, stream>>>(x, pos, xsT);

  // 5) k = wk * xs + bk (f16 [n,C]) ; v = wv * xs + bv (f16 [C,n])
  gemm_wmma_kernel<1><<<gemm_grid, 128, 0, stream>>>(wkh, xsT, bk, nullptr, kT, kC,
                                                     kHW, kC);
  gemm_wmma_kernel<2><<<gemm_grid, 128, 0, stream>>>(wvh, xsT, bv, nullptr, vhd, kC,
                                                     kHW, kC);

  // 6) fused attention
  attn_kernel<<<dim3(kHW / 16, kB * kHEADS), 32, 0, stream>>>(qT, kT, vhd, pos, rpe,
                                                              oT);

  // 7) y = wo * o + bo -> d_out (f32 [C,HW])
  gemm_wmma_kernel<3><<<gemm_grid, 128, 0, stream>>>(woh, oT, bo, y_out, nullptr, kC,
                                                     kHW, kC);
}
